// Moondream3TextMoE_17978733101588
// MI455X (gfx1250) — matrix operand order repaired and checked by reading, not compiled, not run on hardware
//
#include <hip/hip_runtime.h>
#include <hip/hip_bf16.h>
#include <math.h>

// ---------------------------------------------------------------------------
// Moondream3 Text MoE for MI455X (gfx1250).
// All heavy math on v_wmma_f32_16x16x32_bf16 (f32 accumulate, bf16 operands
// converted on the fly). Workspace use ~337 MB:
//   logits, topk ids/weights, slots, counts, xe(bf16), act(bf16), y(bf16).
// ---------------------------------------------------------------------------

#define T_TOK 2048
#define H_DIM 2048
#define I_DIM 1024
#define E_EXP 64
#define TOPK  8
#define CAP   512

#define BM 128
#define BN 128
#define BK 32
#define LDA 40   // padded LDS row stride in bf16 elements (80B)

typedef __bf16 bf16_t;
typedef __bf16 bf16x8 __attribute__((ext_vector_type(8)));
typedef __bf16 v16bf  __attribute__((ext_vector_type(16)));
typedef float  f32x4  __attribute__((ext_vector_type(4)));
typedef float  f32x8  __attribute__((ext_vector_type(8)));
typedef float  v8f    __attribute__((ext_vector_type(8)));

union FragBF { bf16x8 h[2]; v16bf v; };
union FragF  { v8f v; float f[8]; };

// ---------------------------------------------------------------- gate logits
__global__ __launch_bounds__(256) void gate_kernel(
    const float* __restrict__ x, const float* __restrict__ gw,
    const float* __restrict__ gb, float* __restrict__ logits) {
  int tid = threadIdx.x;
  int tok = blockIdx.x * 4 + (tid >> 6);
  int e   = tid & 63;
  const float* xr = x  + (size_t)tok * H_DIM;
  const float* wr = gw + (size_t)e   * H_DIM;
  float acc = 0.f;
  for (int h = 0; h < H_DIM; h += 4) {
    f32x4 xv = *(const f32x4*)(xr + h);
    f32x4 wv = *(const f32x4*)(wr + h);
    acc += xv.x * wv.x + xv.y * wv.y + xv.z * wv.z + xv.w * wv.w;
  }
  logits[tok * E_EXP + e] = acc + gb[e];
}

// ----------------------------------------------------------- top-8 + softmax
__global__ __launch_bounds__(256) void topk_kernel(
    const float* __restrict__ logits, int* __restrict__ ids,
    float* __restrict__ wts) {
  int tok = blockIdx.x * 256 + threadIdx.x;
  if (tok >= T_TOK) return;
  const float* lr = logits + (size_t)tok * E_EXP;
  unsigned long long taken = 0ull;
  float topv[TOPK]; int topi[TOPK];
#pragma unroll
  for (int k = 0; k < TOPK; ++k) {
    float best = -INFINITY; int bi = 0;
    for (int j = 0; j < E_EXP; ++j) {
      float v = lr[j];
      bool freebit = ((taken >> j) & 1ull) == 0ull;
      if (freebit && v > best) { best = v; bi = j; }
    }
    taken |= (1ull << bi);
    topv[k] = best; topi[k] = bi;
  }
  float mx = topv[0], s = 0.f, ev[TOPK];
#pragma unroll
  for (int k = 0; k < TOPK; ++k) { ev[k] = expf(topv[k] - mx); s += ev[k]; }
  float inv = 1.f / s;
#pragma unroll
  for (int k = 0; k < TOPK; ++k) {
    ids[tok * TOPK + k] = topi[k];
    wts[tok * TOPK + k] = ev[k] * inv;
  }
}

// ------------------------------------------- order-preserving capacity route
__global__ __launch_bounds__(64) void route_kernel(
    const int* __restrict__ ids, int* __restrict__ slot, int* __restrict__ cnt) {
  int e = threadIdx.x;   // one thread per expert; flat order preserved
  int c = 0;
  for (int f = 0; f < T_TOK * TOPK; ++f) {
    if (ids[f] == e) {
      slot[f] = (c < CAP) ? (e * CAP + c) : -1;
      ++c;
    }
  }
  cnt[e] = (c < CAP) ? c : CAP;
}

// ------------------------------------------------- scatter x -> xe (bf16)
__global__ __launch_bounds__(256) void scatter_kernel(
    const float* __restrict__ x, const int* __restrict__ slot,
    bf16_t* __restrict__ xe) {
  int f = blockIdx.x;
  int s = slot[f];
  if (s < 0) return;
  int tok = f >> 3;
  const float* src = x + (size_t)tok * H_DIM;
  bf16_t* dst = xe + (size_t)s * H_DIM;
  int h0 = threadIdx.x * 8;
  f32x4 a = *(const f32x4*)(src + h0);
  f32x4 b = *(const f32x4*)(src + h0 + 4);
  f32x8 v = { a.x, a.y, a.z, a.w, b.x, b.y, b.z, b.w };
  *(bf16x8*)(dst + h0) = __builtin_convertvector(v, bf16x8);
}

// -------------------------------------------------------------- GEMM helpers
// A frag (16x32 bf16): lane = m + 16*s holds K = {8s..8s+7, 16+8s..23+8s}
// B frag (32x16 bf16): lane = n + 16*s holds K = {8s..8s+7, 16+8s..23+8s}
__device__ __forceinline__ void load_afrag(const bf16_t* row, int ls, FragBF& fr) {
  fr.h[0] = *(const bf16x8*)(row + 8 * ls);
  fr.h[1] = *(const bf16x8*)(row + 16 + 8 * ls);
}
__device__ __forceinline__ void load_bfrag(const bf16_t* row, int ls, FragBF& fr) {
  fr.h[0] = *(const bf16x8*)(row + 8 * ls);
  fr.h[1] = *(const bf16x8*)(row + 16 + 8 * ls);
}

// --------------------------------------------- GEMM1: h_full + GELU-gate fuse
__global__ __launch_bounds__(256) void gemm1_kernel(
    const bf16_t* __restrict__ xe, const float* __restrict__ fc1,
    const int* __restrict__ cnt, bf16_t* __restrict__ act) {
  const int e  = blockIdx.z;
  const int m0 = blockIdx.y * BM;      // over CAP
  const int n0 = blockIdx.x * BN;      // over I (h cols; g at +I)
  const int tid  = threadIdx.x;
  const int lane = tid & 31;
  const int w    = tid >> 5;
  const int wm   = (w >> 2) * 64;
  const int wn   = (w & 3) * 32;
  const int ln   = lane & 15;
  const int ls   = lane >> 4;
  const int rows_valid = cnt[e];

  __shared__ __align__(16) bf16_t sA [BM * LDA];
  __shared__ __align__(16) bf16_t sBh[BN * LDA];
  __shared__ __align__(16) bf16_t sBg[BN * LDA];

  const bf16_t* Ab = xe + (size_t)e * CAP * H_DIM;
  const float*  Bh = fc1 + ((size_t)e * (2 * I_DIM) + n0) * H_DIM;
  const float*  Bg = Bh + (size_t)I_DIM * H_DIM;

  const int lr = tid >> 1;   // tile row 0..127
  const int lp = tid & 1;    // 16-element half of the 32-wide K slab

  const bf16x8 bz = {};
  bf16x8 aReg[2];
  f32x8  bhReg[2], bgReg[2];

  auto loadA = [&](int k0) {
    int gr = m0 + lr;
    if (gr < rows_valid) {
      const bf16_t* p = Ab + (size_t)gr * H_DIM + k0 + lp * 16;
      aReg[0] = *(const bf16x8*)(p);
      aReg[1] = *(const bf16x8*)(p + 8);
    } else { aReg[0] = bz; aReg[1] = bz; }
  };
  auto loadB = [&](const float* base, int k0, f32x8* r) {
    const float* p = base + (size_t)lr * H_DIM + k0 + lp * 16;
    r[0] = *(const f32x8*)(p);
    r[1] = *(const f32x8*)(p + 8);
  };
  auto stage = [&]() {
    bf16_t* pa = sA + lr * LDA + lp * 16;
    *(bf16x8*)(pa)     = aReg[0];
    *(bf16x8*)(pa + 8) = aReg[1];
    bf16_t* ph = sBh + lr * LDA + lp * 16;
    *(bf16x8*)(ph)     = __builtin_convertvector(bhReg[0], bf16x8);
    *(bf16x8*)(ph + 8) = __builtin_convertvector(bhReg[1], bf16x8);
    bf16_t* pg = sBg + lr * LDA + lp * 16;
    *(bf16x8*)(pg)     = __builtin_convertvector(bgReg[0], bf16x8);
    *(bf16x8*)(pg + 8) = __builtin_convertvector(bgReg[1], bf16x8);
  };

  v8f ch[4][2] = {}; v8f cg[4][2] = {};

  loadA(0); loadB(Bh, 0, bhReg); loadB(Bg, 0, bgReg);
  for (int k0 = 0; k0 < H_DIM; k0 += BK) {
    __syncthreads();          // prior compute done reading LDS
    stage();
    __syncthreads();
    if (k0 + BK < H_DIM) {    // prefetch next slab while computing
      loadA(k0 + BK); loadB(Bh, k0 + BK, bhReg); loadB(Bg, k0 + BK, bgReg);
    }
    FragBF af[4], bhf[2], bgf[2];
#pragma unroll
    for (int i = 0; i < 4; ++i)
      load_afrag(sA + (wm + i * 16 + ln) * LDA, ls, af[i]);
#pragma unroll
    for (int j = 0; j < 2; ++j) {
      load_bfrag(sBh + (wn + j * 16 + ln) * LDA, ls, bhf[j]);
      load_bfrag(sBg + (wn + j * 16 + ln) * LDA, ls, bgf[j]);
    }
#pragma unroll
    for (int i = 0; i < 4; ++i)
#pragma unroll
      for (int j = 0; j < 2; ++j) {
        ch[i][j] = __builtin_amdgcn_wmma_f32_16x16x32_bf16(
            false, af[i].v, false, bhf[j].v, (short)0, ch[i][j], false, false);
        cg[i][j] = __builtin_amdgcn_wmma_f32_16x16x32_bf16(
            false, af[i].v, false, bgf[j].v, (short)0, cg[i][j], false, false);
      }
  }

  // epilogue: act = gelu_exact(h) * (g + 1)
  const size_t actBase = (size_t)e * CAP * I_DIM;
#pragma unroll
  for (int i = 0; i < 4; ++i)
#pragma unroll
    for (int j = 0; j < 2; ++j) {
      FragF fh, fg; fh.v = ch[i][j]; fg.v = cg[i][j];
#pragma unroll
      for (int r = 0; r < 8; ++r) {
        int row = m0 + wm + i * 16 + ls * 8 + r;
        int col = n0 + wn + j * 16 + ln;
        float hv = fh.f[r], gv = fg.f[r];
        float a = 0.5f * hv * (1.0f + erff(hv * 0.70710678118654752f));
        act[actBase + (size_t)row * I_DIM + col] = (bf16_t)(a * (gv + 1.0f));
      }
    }
}

// ------------------------------------------------------------ GEMM2: y = act@fc2^T
__global__ __launch_bounds__(256) void gemm2_kernel(
    const bf16_t* __restrict__ act, const float* __restrict__ fc2,
    bf16_t* __restrict__ y) {
  const int e  = blockIdx.z;
  const int m0 = blockIdx.y * BM;      // over CAP
  const int n0 = blockIdx.x * BN;      // over H
  const int tid  = threadIdx.x;
  const int lane = tid & 31;
  const int w    = tid >> 5;
  const int wm   = (w >> 2) * 64;
  const int wn   = (w & 3) * 32;
  const int ln   = lane & 15;
  const int ls   = lane >> 4;

  __shared__ __align__(16) bf16_t sA[BM * LDA];
  __shared__ __align__(16) bf16_t sB[BN * LDA];

  const bf16_t* Ab = act + (size_t)e * CAP * I_DIM;
  const float*  Bb = fc2 + ((size_t)e * H_DIM + n0) * I_DIM;

  const int lr = tid >> 1;
  const int lp = tid & 1;

  bf16x8 aReg[2];
  f32x8  bReg[2];

  auto loadA = [&](int k0) {
    const bf16_t* p = Ab + (size_t)(m0 + lr) * I_DIM + k0 + lp * 16;
    aReg[0] = *(const bf16x8*)(p);
    aReg[1] = *(const bf16x8*)(p + 8);
  };
  auto loadB = [&](int k0) {
    const float* p = Bb + (size_t)lr * I_DIM + k0 + lp * 16;
    bReg[0] = *(const f32x8*)(p);
    bReg[1] = *(const f32x8*)(p + 8);
  };
  auto stage = [&]() {
    bf16_t* pa = sA + lr * LDA + lp * 16;
    *(bf16x8*)(pa)     = aReg[0];
    *(bf16x8*)(pa + 8) = aReg[1];
    bf16_t* pb = sB + lr * LDA + lp * 16;
    *(bf16x8*)(pb)     = __builtin_convertvector(bReg[0], bf16x8);
    *(bf16x8*)(pb + 8) = __builtin_convertvector(bReg[1], bf16x8);
  };

  v8f c[4][2] = {};

  loadA(0); loadB(0);
  for (int k0 = 0; k0 < I_DIM; k0 += BK) {
    __syncthreads();
    stage();
    __syncthreads();
    if (k0 + BK < I_DIM) { loadA(k0 + BK); loadB(k0 + BK); }
    FragBF af[4], bf[2];
#pragma unroll
    for (int i = 0; i < 4; ++i)
      load_afrag(sA + (wm + i * 16 + ln) * LDA, ls, af[i]);
#pragma unroll
    for (int j = 0; j < 2; ++j)
      load_bfrag(sB + (wn + j * 16 + ln) * LDA, ls, bf[j]);
#pragma unroll
    for (int i = 0; i < 4; ++i)
#pragma unroll
      for (int j = 0; j < 2; ++j)
        c[i][j] = __builtin_amdgcn_wmma_f32_16x16x32_bf16(
            false, af[i].v, false, bf[j].v, (short)0, c[i][j], false, false);
  }

  const size_t yBase = (size_t)e * CAP * H_DIM;
#pragma unroll
  for (int i = 0; i < 4; ++i)
#pragma unroll
    for (int j = 0; j < 2; ++j) {
      FragF fc; fc.v = c[i][j];
#pragma unroll
      for (int r = 0; r < 8; ++r) {
        int row = m0 + wm + i * 16 + ls * 8 + r;
        int col = n0 + wn + j * 16 + ln;
        y[yBase + (size_t)row * H_DIM + col] = (bf16_t)fc.f[r];
      }
    }
}

// ----------------------------------------------------- weighted gather to out
__global__ __launch_bounds__(256) void gather_kernel(
    const bf16_t* __restrict__ y, const int* __restrict__ slot,
    const float* __restrict__ wts, float* __restrict__ out) {
  int tok = blockIdx.x;
  int h0  = threadIdx.x * 8;
  float acc[8] = {0.f, 0.f, 0.f, 0.f, 0.f, 0.f, 0.f, 0.f};
#pragma unroll
  for (int k = 0; k < TOPK; ++k) {
    int s = slot[tok * TOPK + k];
    float wgt = wts[tok * TOPK + k];
    if (s >= 0) {
      bf16x8 v = *(const bf16x8*)(y + (size_t)s * H_DIM + h0);
      f32x8 f = __builtin_convertvector(v, f32x8);
#pragma unroll
      for (int q = 0; q < 8; ++q) acc[q] += wgt * f[q];
    }
  }
  f32x4 o0 = { acc[0], acc[1], acc[2], acc[3] };
  f32x4 o1 = { acc[4], acc[5], acc[6], acc[7] };
  *(f32x4*)(out + (size_t)tok * H_DIM + h0)     = o0;
  *(f32x4*)(out + (size_t)tok * H_DIM + h0 + 4) = o1;
}

// ---------------------------------------------------------------------------
extern "C" void kernel_launch(void* const* d_in, const int* in_sizes, int n_in,
                              void* d_out, int out_size, void* d_ws, size_t ws_size,
                              hipStream_t stream) {
  (void)in_sizes; (void)n_in; (void)out_size; (void)ws_size;
  const float* x   = (const float*)d_in[0];
  const float* gw  = (const float*)d_in[1];
  const float* gb  = (const float*)d_in[2];
  const float* fc1 = (const float*)d_in[3];
  const float* fc2 = (const float*)d_in[4];
  float* out = (float*)d_out;

  char* ws = (char*)d_ws;
  size_t off = 0;
  auto alloc = [&](size_t bytes) {
    void* p = ws + off;
    off += (bytes + 255) & ~(size_t)255;
    return p;
  };
  float*  logits = (float*) alloc((size_t)T_TOK * E_EXP * 4);
  int*    ids    = (int*)   alloc((size_t)T_TOK * TOPK * 4);
  float*  wts    = (float*) alloc((size_t)T_TOK * TOPK * 4);
  int*    slot   = (int*)   alloc((size_t)T_TOK * TOPK * 4);
  int*    cnt    = (int*)   alloc((size_t)E_EXP * 4);
  bf16_t* xe     = (bf16_t*)alloc((size_t)E_EXP * CAP * H_DIM * 2);
  bf16_t* act    = (bf16_t*)alloc((size_t)E_EXP * CAP * I_DIM * 2);
  bf16_t* yb     = (bf16_t*)alloc((size_t)E_EXP * CAP * H_DIM * 2);

  gate_kernel  <<<T_TOK / 4,   256, 0, stream>>>(x, gw, gb, logits);
  topk_kernel  <<<T_TOK / 256, 256, 0, stream>>>(logits, ids, wts);
  route_kernel <<<1,           64,  0, stream>>>(ids, slot, cnt);
  scatter_kernel<<<T_TOK * TOPK, 256, 0, stream>>>(x, slot, xe);
  gemm1_kernel <<<dim3(I_DIM / BN, CAP / BM, E_EXP), 256, 0, stream>>>(xe, fc1, cnt, act);
  gemm2_kernel <<<dim3(H_DIM / BN, CAP / BM, E_EXP), 256, 0, stream>>>(act, fc2, yb);
  gather_kernel<<<T_TOK,       256, 0, stream>>>(yb, slot, wts, out);
}
